// PositionSensitiveKeywordAttention_33895881900739
// MI455X (gfx1250) — matrix operand order repaired and checked by reading, not compiled, not run on hardware
//
#include <hip/hip_runtime.h>
#include <hip/hip_bf16.h>
#include <math.h>

typedef __bf16 bf16_t;
typedef __attribute__((ext_vector_type(16))) __bf16 v16bf;
typedef __attribute__((ext_vector_type(8)))  __bf16 v8bf;
typedef __attribute__((ext_vector_type(8)))  float  v8f;

#define DEV __device__ __forceinline__

DEV v8f wmma_bf16(v16bf a, v16bf b, v8f c) {
  // D = A(16x32) * B(32x16) + C, f32 accumulate
  return __builtin_amdgcn_wmma_f32_16x16x32_bf16(false, a, false, b, (short)0, c, false, false);
}

// A fragment 16x32 from row-major [M x K] slab (base pre-offset to (row0,k0)).
// Lane m = lane&15, half kh = lane>>4 holds K chunks [kh*8, kh*8+8) and [16+kh*8, 16+kh*8+8).
DEV v16bf load_a_bf16(const bf16_t* A, int lda, int lane) {
  const int m = lane & 15, kh = lane >> 4;
  const bf16_t* p = A + (size_t)m * lda + kh * 8;
  union { v16bf v; v8bf h[2]; } u;
  u.h[0] = *(const v8bf*)(p);
  u.h[1] = *(const v8bf*)(p + 16);
  return u.v;
}

DEV v16bf load_a_f32(const float* A, int lda, int lane) {
  const int m = lane & 15, kh = lane >> 4;
  const float* p = A + (size_t)m * lda + kh * 8;
  v16bf r;
#pragma unroll
  for (int j = 0; j < 8; ++j) { r[j] = (bf16_t)p[j]; r[j + 8] = (bf16_t)p[16 + j]; }
  return r;
}

// B fragment 32x16 from an N-major [N x K] slab (base pre-offset to (n0,k0)).
// Lane n = lane&15, half kg = lane>>4 holds K chunks [kg*8, kg*8+8) and [16+kg*8, 16+kg*8+8).
DEV v16bf load_bt(const bf16_t* Bt, int ldbt, int lane) {
  const int n = lane & 15, kg = lane >> 4;
  const bf16_t* p = Bt + (size_t)n * ldbt + kg * 8;
  union { v16bf v; v8bf h[2]; } u;
  u.h[0] = *(const v8bf*)(p);
  u.h[1] = *(const v8bf*)(p + 16);
  return u.v;
}

// prefetch the same lane's B-fragment row, `dist` elements further in K
DEV void pf_bt(const bf16_t* Bt, int ldbt, int lane, int dist) {
  const int n = lane & 15, kg = lane >> 4;
  __builtin_prefetch((const void*)(Bt + (size_t)n * ldbt + kg * 16 + dist), 0, 0);
}

// wave32 butterfly reductions (all lanes end with the result)
DEV float wred_max(float v) {
#pragma unroll
  for (int off = 16; off; off >>= 1) v = fmaxf(v, __shfl_xor(v, off, 32));
  return v;
}
DEV float wred_sum(float v) {
#pragma unroll
  for (int off = 16; off; off >>= 1) v += __shfl_xor(v, off, 32);
  return v;
}

// ---------------- generic WMMA GEMM: C = act(A @ Bt^T + bias) ----------------
// block = 256 threads = 8 waves; wave computes a 16(M) x 64(N) strip.
template<int ACT, int AF32, int OUTF32>
__global__ __launch_bounds__(256)
void gemm_wmma(const void* __restrict__ Ain, const bf16_t* __restrict__ Bt,
               const float* __restrict__ bias, void* __restrict__ Cout,
               int M, int N, int K, int lda, int ldbt, int ldc,
               long azs, long bzs, long czs)
{
  const int lane = threadIdx.x & 31;
  const int wave = threadIdx.x >> 5;
  const int row0 = blockIdx.x * 128 + wave * 16;
  const int n0   = blockIdx.y * 64;
  if (row0 >= M || n0 >= N) return;

  v8f acc[4] = {};
  const bf16_t* B = Bt + (size_t)blockIdx.z * bzs;

  for (int k0 = 0; k0 < K; k0 += 32) {
    v16bf a;
    if constexpr (AF32) {
      const float* A = (const float*)Ain + (size_t)blockIdx.z * azs;
      a = load_a_f32(A + (size_t)row0 * lda + k0, lda, lane);
    } else {
      const bf16_t* A = (const bf16_t*)Ain + (size_t)blockIdx.z * azs;
      a = load_a_bf16(A + (size_t)row0 * lda + k0, lda, lane);
    }
    const bool pf = (k0 + 256) < K;
#pragma unroll
    for (int j = 0; j < 4; ++j) {
      const bf16_t* Bj = B + (size_t)(n0 + j * 16) * ldbt + k0;
      v16bf b = load_bt(Bj, ldbt, lane);
      if (pf) pf_bt(Bj, ldbt, lane, 256);   // global_prefetch_b8, 8 K-steps ahead
      acc[j] = wmma_bf16(a, b, acc[j]);
    }
  }

  const int n = lane & 15, mh = lane >> 4;
#pragma unroll
  for (int j = 0; j < 4; ++j) {
    const int col = n0 + j * 16 + n;
    const float bv = bias ? bias[col] : 0.0f;
#pragma unroll
    for (int r = 0; r < 8; ++r) {
      float v = acc[j][r] + bv;
      if constexpr (ACT == 1) v = fmaxf(v, 0.0f);
      if constexpr (ACT == 2) v = tanhf(v);
      const size_t idx = (size_t)blockIdx.z * czs + (size_t)(row0 + mh * 8 + r) * ldc + col;
      if constexpr (OUTF32) ((float*)Cout)[idx] = v;
      else                  ((bf16_t*)Cout)[idx] = (bf16_t)v;
    }
  }
}

// ---------------- fused scores -> softmax -> head-mean -> dist-softmax ----------------
#define SKLEN 2048
#define SQLEN 2048
#define DDIM  512
#define NHEAD 8
#define NTHR  1024   // 32 waves: one (LDS-limited) workgroup fills the WGP

__global__ __launch_bounds__(NTHR)
void attn_kernel(const bf16_t* __restrict__ qp, const bf16_t* __restrict__ kp,
                 const int* __restrict__ kwpos, float* __restrict__ adjusted)
{
  __shared__ float srow[16][SKLEN];   // per-head scores / scratch (128 KB)
  __shared__ float accs[16][SKLEN];   // head-mean accumulator (128 KB)
  __shared__ float red[16][2];        // two wave-partials per q-row
  __shared__ int   kposs[SKLEN];

  const int q0  = blockIdx.x * 16;
  const int b   = blockIdx.y;
  const int tid = threadIdx.x;
  const int lane = tid & 31, wave = tid >> 5;        // wave 0..31

  for (int i = tid; i < SKLEN; i += NTHR) kposs[i] = kwpos[(size_t)b * SKLEN + i];
  for (int i = tid; i < 16 * SKLEN; i += NTHR) (&accs[0][0])[i] = 0.0f;
  __syncthreads();

  // 64 threads per q-row; each wave's 32 lanes all belong to one row.
  const int row = tid >> 6, sub = tid & 63, half = sub >> 5;

  for (int h = 0; h < NHEAD; ++h) {
    const bf16_t* qb = qp + ((size_t)b * SQLEN + q0) * DDIM + h * 64;
    const bf16_t* kb = kp + (size_t)b * SKLEN * DDIM + h * 64;

    // scores: 128 k-tiles over 32 waves -> exactly 4 tiles/wave,
    // kept as 4 independent accumulator chains for WMMA ILP.
    {
      v8f s[4] = {};
      v16bf a0 = load_a_bf16(qb +  0, DDIM, lane);
      v16bf a1 = load_a_bf16(qb + 32, DDIM, lane);
#pragma unroll
      for (int j = 0; j < 4; ++j) {
        const int kt = wave + j * 32;
        const bf16_t* kbj = kb + (size_t)kt * 16 * DDIM;
        v16bf b0 = load_bt(kbj +  0, DDIM, lane);
        v16bf b1 = load_bt(kbj + 32, DDIM, lane);
        s[j] = wmma_bf16(a0, b0, s[j]);
        s[j] = wmma_bf16(a1, b1, s[j]);
      }
      const int n = lane & 15, mh = lane >> 4;
#pragma unroll
      for (int j = 0; j < 4; ++j) {
        const int c0 = (wave + j * 32) * 16 + n;
#pragma unroll
        for (int r = 0; r < 8; ++r) srow[mh * 8 + r][c0] = s[j][r] * 0.125f;
      }
    }
    __syncthreads();

    // rowwise softmax: strided scan + wave shfl-reduce + 2-way LDS combine
    float m = -3.0e38f;
    for (int c = sub; c < SKLEN; c += 64) m = fmaxf(m, srow[row][c]);
    m = wred_max(m);
    if (lane == 0) red[row][half] = m;
    __syncthreads();
    m = fmaxf(red[row][0], red[row][1]);
    __syncthreads();
    float sum = 0.0f;
    for (int c = sub; c < SKLEN; c += 64) {
      float e = __expf(srow[row][c] - m);
      srow[row][c] = e;
      sum += e;
    }
    sum = wred_sum(sum);
    if (lane == 0) red[row][half] = sum;
    __syncthreads();
    sum = red[row][0] + red[row][1];
    const float inv = 0.125f / sum;   // includes 1/H for the head mean
    for (int c = sub; c < SKLEN; c += 64) accs[row][c] += srow[row][c] * inv;
    __syncthreads();
  }

  // adjusted = softmax(accs * 1/(1+|qpos-kpos|))
  float m = -3.0e38f;
  for (int c = sub; c < SKLEN; c += 64) {
    float d = 1.0f / (1.0f + fabsf((float)(q0 + row - kposs[c])));
    float v = accs[row][c] * d;
    srow[row][c] = v;
    m = fmaxf(m, v);
  }
  m = wred_max(m);
  if (lane == 0) red[row][half] = m;
  __syncthreads();
  m = fmaxf(red[row][0], red[row][1]);
  __syncthreads();
  float sum = 0.0f;
  for (int c = sub; c < SKLEN; c += 64) {
    float e = __expf(srow[row][c] - m);
    srow[row][c] = e;
    sum += e;
  }
  sum = wred_sum(sum);
  if (lane == 0) red[row][half] = sum;
  __syncthreads();
  sum = red[row][0] + red[row][1];
  const float inv = 1.0f / sum;
  float* outrow = adjusted + ((size_t)b * SQLEN + q0 + row) * SKLEN;
  for (int c = sub; c < SKLEN; c += 64) outrow[c] = srow[row][c] * inv;
}

// ---------------- small prep kernels ----------------
__global__ void cvt_bf16_kernel(const float* __restrict__ src, bf16_t* __restrict__ dst, size_t n) {
  size_t i = (size_t)blockIdx.x * 256 + threadIdx.x;
  if (i < n) dst[i] = (bf16_t)src[i];
}

// dst[z][c][r] = (bf16) src[z][r][c]   (src R x C row-major -> dst C x R row-major)
__global__ void transpose_bf16_kernel(const float* __restrict__ src, bf16_t* __restrict__ dst,
                                      int R, int C, long szs, long dzs) {
  const int c = blockIdx.x * 16 + threadIdx.x;
  const int r = blockIdx.y * 16 + threadIdx.y;
  if (r < R && c < C)
    dst[(size_t)blockIdx.z * dzs + (size_t)c * R + r] =
        (bf16_t)src[(size_t)blockIdx.z * szs + (size_t)r * C + c];
}

// kp[row][c] = c<512 ? key[row][c] : pos_emb[kwpos[row]][c-512]
__global__ void build_kp_kernel(const float* __restrict__ key, const int* __restrict__ kwpos,
                                const float* __restrict__ pe, bf16_t* __restrict__ kp, size_t total) {
  size_t i = (size_t)blockIdx.x * 256 + threadIdx.x;
  if (i >= total) return;
  const int c = (int)(i % 576);
  const size_t row = i / 576;
  float v;
  if (c < 512) v = key[row * 512 + c];
  else         v = pe[(size_t)kwpos[row] * 64 + (c - 512)];
  kp[i] = (bf16_t)v;
}

extern "C" void kernel_launch(void* const* d_in, const int* in_sizes, int n_in,
                              void* d_out, int out_size, void* d_ws, size_t ws_size,
                              hipStream_t stream) {
  const float* query   = (const float*)d_in[0];
  const float* key     = (const float*)d_in[1];
  const float* value   = (const float*)d_in[2];
  const int*   kwpos   = (const int*)  d_in[3];
  const float* pos_emb = (const float*)d_in[4];
  const float* w1 = (const float*)d_in[5];
  const float* b1 = (const float*)d_in[6];
  const float* w2 = (const float*)d_in[7];
  const float* b2 = (const float*)d_in[8];
  const float* wq = (const float*)d_in[9];
  const float* bq = (const float*)d_in[10];
  const float* wk = (const float*)d_in[11];
  const float* bk = (const float*)d_in[12];
  (void)in_sizes; (void)n_in; (void)out_size; (void)ws_size;

  const int B = 4, Sq = 2048, Sk = 2048, D = 512, KP1 = 576;

  char* ws = (char*)d_ws;
  size_t ofs = 0;
  auto alloc = [&](size_t bytes) { void* p = ws + ofs; ofs += (bytes + 255) & ~(size_t)255; return p; };
  bf16_t* W1T = (bf16_t*)alloc((size_t)D * KP1 * 2);
  bf16_t* W2T = (bf16_t*)alloc((size_t)D * D * 2);
  bf16_t* WQT = (bf16_t*)alloc((size_t)D * D * 2);
  bf16_t* WKT = (bf16_t*)alloc((size_t)D * D * 2);
  bf16_t* KPB = (bf16_t*)alloc((size_t)B * Sk * KP1 * 2);  // [key | pe] bf16
  bf16_t* HB  = (bf16_t*)alloc((size_t)B * Sk * D * 2);    // relu hidden
  bf16_t* EKB = (bf16_t*)alloc((size_t)B * Sk * D * 2);    // enhanced key
  bf16_t* QB  = (bf16_t*)alloc((size_t)B * Sq * D * 2);    // query bf16
  bf16_t* QPB = (bf16_t*)alloc((size_t)B * Sq * D * 2);    // q projection
  bf16_t* KPJ = (bf16_t*)alloc((size_t)B * Sk * D * 2);    // k projection
  bf16_t* VTB = (bf16_t*)alloc((size_t)B * D * Sk * 2);    // value^T per batch

  float* out_main = (float*)d_out;                     // (B,Sq,D)
  float* adjusted = out_main + (size_t)B * Sq * D;     // (B,Sq,Sk)

  dim3 t16(16, 16, 1);
  // weights -> bf16, N-major
  transpose_bf16_kernel<<<dim3(D / 16, KP1 / 16, 1), t16, 0, stream>>>(w1, W1T, KP1, D, 0, 0);
  transpose_bf16_kernel<<<dim3(D / 16, D / 16, 1),  t16, 0, stream>>>(w2, W2T, D, D, 0, 0);
  transpose_bf16_kernel<<<dim3(D / 16, D / 16, 1),  t16, 0, stream>>>(wq, WQT, D, D, 0, 0);
  transpose_bf16_kernel<<<dim3(D / 16, D / 16, 1),  t16, 0, stream>>>(wk, WKT, D, D, 0, 0);
  // value -> bf16 transposed per batch: (Sk x D) -> (D x Sk)
  transpose_bf16_kernel<<<dim3(D / 16, Sk / 16, B), t16, 0, stream>>>(value, VTB, Sk, D,
                                                                     (long)Sk * D, (long)D * Sk);
  size_t nkp = (size_t)B * Sk * KP1;
  build_kp_kernel<<<(unsigned)((nkp + 255) / 256), 256, 0, stream>>>(key, kwpos, pos_emb, KPB, nkp);
  size_t nq = (size_t)B * Sq * D;
  cvt_bf16_kernel<<<(unsigned)((nq + 255) / 256), 256, 0, stream>>>(query, QB, nq);

  const int M = B * Sk;  // 8192
  // MLP1: h = relu(kp @ w1 + b1)
  gemm_wmma<1, 0, 0><<<dim3(M / 128, D / 64, 1), 256, 0, stream>>>(
      KPB, W1T, b1, HB, M, D, KP1, KP1, KP1, D, 0, 0, 0);
  // MLP2: ek = tanh(h @ w2 + b2)
  gemm_wmma<2, 0, 0><<<dim3(M / 128, D / 64, 1), 256, 0, stream>>>(
      HB, W2T, b2, EKB, M, D, D, D, D, D, 0, 0, 0);
  // q = query @ wq + bq
  gemm_wmma<0, 0, 0><<<dim3(M / 128, D / 64, 1), 256, 0, stream>>>(
      QB, WQT, bq, QPB, M, D, D, D, D, D, 0, 0, 0);
  // k = ek @ wk + bk
  gemm_wmma<0, 0, 0><<<dim3(M / 128, D / 64, 1), 256, 0, stream>>>(
      EKB, WKT, bk, KPJ, M, D, D, D, D, D, 0, 0, 0);

  // fused attention chain -> adjusted
  attn_kernel<<<dim3(Sq / 16, B, 1), NTHR, 0, stream>>>(QPB, KPJ, kwpos, adjusted);

  // out = adjusted @ value  (A is f32, converted to bf16 on the fly)
  gemm_wmma<0, 1, 1><<<dim3(Sq / 128, D / 64, B), 256, 0, stream>>>(
      adjusted, VTB, (const float*)nullptr, out_main,
      Sq, D, Sk, Sk, Sk, D, (long)Sq * Sk, (long)D * Sk, (long)Sq * D);
}
